// DRNN_Char_45801531244638
// MI455X (gfx1250) — matrix operand order repaired and checked by reading, not compiled, not run on hardware
//
#include <hip/hip_runtime.h>
#include <hip/hip_bf16.h>

#define EMB 256
#define HID 512
#define NVOCAB 256
#define BB 16
#define TT 2048
#define MROWS (BB*TT)       // 32768 rows (time-major: row = t*B + b)
#define G3 (3*HID)          // 1536 gate columns

typedef __attribute__((ext_vector_type(16))) __bf16 v16bf;
typedef __attribute__((ext_vector_type(8)))  float  v8f;
typedef __attribute__((ext_vector_type(8)))  int    v8i;
typedef __attribute__((ext_vector_type(4)))  int    v4i;

#if defined(__has_builtin)
#  if __has_builtin(__builtin_amdgcn_global_load_async_to_lds_b128) && \
      __has_builtin(__builtin_amdgcn_s_wait_asynccnt)
#    define ASYNC_LDS 1
#  endif
#endif
#ifndef ASYNC_LDS
#  define ASYNC_LDS 0
#endif

#if ASYNC_LDS
typedef __attribute__((address_space(1))) v4i* gptr_v4i;   // global
typedef __attribute__((address_space(3))) v4i* lptr_v4i;   // LDS
__device__ __forceinline__ void async_tile16(const __hip_bfloat16* gsrc,
                                             __hip_bfloat16* ldst) {
  __builtin_amdgcn_global_load_async_to_lds_b128(
      (gptr_v4i)(v4i*)(__hip_bfloat16*)gsrc, (lptr_v4i)(v4i*)ldst, 0, 0);
}
#endif

// ------------------------------------------------------------------
// Embedding gather: X0[t*B + b, e] = bf16(emb[x[b,t], e])
// ------------------------------------------------------------------
__global__ __launch_bounds__(256) void embed_k(const int* __restrict__ x,
                                               const float* __restrict__ emb,
                                               __hip_bfloat16* __restrict__ X0) {
  int idx = blockIdx.x * 256 + threadIdx.x;   // over MROWS*EMB (exact multiple)
  int row = idx >> 8;                          // EMB == 256
  int e   = idx & 255;
  int t   = row >> 4;                          // BB == 16
  int b   = row & 15;
  int tok = x[b * TT + t];
  X0[idx] = __float2bfloat16(emb[tok * EMB + e]);
}

// ------------------------------------------------------------------
// Pack f32 weights (K x N, row-major) into WMMA B-fragment order:
// for each (kTile of 32, nTile of 16): 32 lanes x 16 contiguous bf16.
// lane l: col = nTile*16 + (l&15); K = kTile*32 + (l>=16 ? 16 : 0) + i
// ------------------------------------------------------------------
__global__ __launch_bounds__(256) void packW_k(const float* __restrict__ W,
                                               __hip_bfloat16* __restrict__ Wp,
                                               int K, int N) {
  int idx  = blockIdx.x * 256 + threadIdx.x;   // over K*N (exact multiple)
  int i    = idx & 15;
  int l    = (idx >> 4) & 31;
  int tile = idx >> 9;
  int nT   = N >> 4;
  int nt   = tile % nT;
  int kt   = tile / nT;
  int col  = nt * 16 + (l & 15);
  int k    = kt * 32 + (l & 16) + i;
  Wp[idx] = __float2bfloat16(W[(size_t)k * N + col]);
}

// ------------------------------------------------------------------
// WMMA GEMM: Y = act(X @ W + bias)
// Block 256 threads (8 waves); block tile 64M x 128N; wave w owns N-tile
// (blockIdx.x*8+w) and all 4 M-subtiles -> 4 WMMAs share one B fragment.
// A tiles (64x32 bf16) double-buffered in LDS; next tile staged during
// compute via GLOBAL_LOAD_ASYNC_TO_LDS (ASYNCcnt) when available.
//   MODE 0: gate epilogue -> bf16 gates[row*G3 + col], tanh(z)/sigm(f,o)
//   MODE 1: decoder epilogue -> f32 out[b*T*V + t*V + col]
// ------------------------------------------------------------------
template <int KDIM, int NDIM, int MODE>
__global__ __launch_bounds__(256) void gemm_wmma(const __hip_bfloat16* __restrict__ X,
                                                 const __hip_bfloat16* __restrict__ Wp,
                                                 const float* __restrict__ bias,
                                                 void* __restrict__ outp) {
  __shared__ __align__(16) __hip_bfloat16 As[2][64 * 32];

  const int tid   = threadIdx.x;
  const int lane  = tid & 31;
  const int wave  = tid >> 5;
  const int nTile = blockIdx.x * 8 + wave;
  const int row0  = blockIdx.y * 64;
  const int nT    = NDIM >> 4;
  const int nK    = KDIM / 32;

  // cooperative A staging: 256 threads x 16B = 4KB tile (64 rows x 32 K)
  const int sRow = tid >> 2;                  // 0..63
  const int sK   = (tid & 3) * 8;             // 0,8,16,24
  const __hip_bfloat16* aSrc = X + (size_t)(row0 + sRow) * KDIM + sK;
  __hip_bfloat16* dstBuf[2] = { &As[0][sRow * 32 + sK], &As[1][sRow * 32 + sK] };

  // per-lane A fragment coords (contiguous k order: element s holds k = kb + s, as in packed B)
  const int arow = lane & 15;
  const int kb   = (lane >> 4) << 4;          // 0 or 16

  v8f acc[4] = {{}, {}, {}, {}};

#if ASYNC_LDS
  async_tile16(aSrc, dstBuf[0]);
  __builtin_amdgcn_s_wait_asynccnt(0);
  __syncthreads();
#else
  *(v4i*)dstBuf[0] = *(const v4i*)aSrc;
  __syncthreads();
#endif

  for (int kt = 0; kt < nK; ++kt) {
    const int cur = kt & 1;
    const __hip_bfloat16* cbuf = As[cur];

#if ASYNC_LDS
    if (kt + 1 < nK)
      async_tile16(aSrc + (size_t)(kt + 1) * 32, dstBuf[1 - cur]);
#else
    v4i treg;
    if (kt + 1 < nK) treg = *(const v4i*)(aSrc + (size_t)(kt + 1) * 32);
#endif

    // B fragment: one contiguous 32B global load from packed weights
    union { v8i i; v16bf v; } bf;
    bf.i = *(const v8i*)(Wp + ((size_t)(kt * nT + nTile) * 32 + lane) * 16);
    if (kt + 1 < nK)
      __builtin_prefetch(Wp + ((size_t)((kt + 1) * nT + nTile) * 32 + lane) * 16, 0, 1);

#pragma unroll
    for (int m = 0; m < 4; ++m) {
      union { v4i h[2]; v16bf v; } af;
      const __hip_bfloat16* ab = &cbuf[(m * 16 + arow) * 32 + kb];
      af.h[0] = *(const v4i*)ab;
      af.h[1] = *(const v4i*)(ab + 8);
      acc[m] = __builtin_amdgcn_wmma_f32_16x16x32_bf16(
          false, af.v, false, bf.v, (short)0, acc[m], false, false);
    }

    if (kt + 1 < nK) {
#if ASYNC_LDS
      __builtin_amdgcn_s_wait_asynccnt(0);
#else
      *(v4i*)dstBuf[1 - cur] = treg;
#endif
      __syncthreads();
    }
  }

  // epilogue (ISA f32 C/D layout: VGPR r -> M=r (lanes 0-15) / M=8+r (16-31))
  const int n     = lane & 15;
  const int col   = nTile * 16 + n;
  const int mbase = (lane >> 4) << 3;
  const float bvv = bias[col];
#pragma unroll
  for (int m = 0; m < 4; ++m) {
#pragma unroll
    for (int r = 0; r < 8; ++r) {
      int row = row0 + m * 16 + mbase + r;
      float y = acc[m][r] + bvv;
      if (MODE == 0) {
        float g = (col < HID) ? tanhf(y) : (1.0f / (1.0f + __expf(-y)));
        ((__hip_bfloat16*)outp)[(size_t)row * G3 + col] = __float2bfloat16(g);
      } else {
        int t = row >> 4;           // row = t*B + b, BB == 16
        int b = row & 15;
        ((float*)outp)[((size_t)b * TT + t) * NVOCAB + col] = y;
      }
    }
  }
}

// ------------------------------------------------------------------
// Dilated fo-pool scan + output gate:
//   c_t = f_t * c_{t-r} + (1 - f_t) * z_t  (per dilation phase j)
//   Xn[row, h] = o_t * c_t  (bf16, next layer input)
// One thread per (j, b, h); gates already activated (tanh/sigmoid).
// ------------------------------------------------------------------
__global__ __launch_bounds__(256) void scan_k(const __hip_bfloat16* __restrict__ G,
                                              __hip_bfloat16* __restrict__ Xn,
                                              int rate) {
  int idx = blockIdx.x * 256 + threadIdx.x;    // over rate*BB*HID (exact multiple)
  int h   = idx % HID;
  int rb  = idx / HID;                         // j*B + b
  int b   = rb & 15;
  int j   = rb >> 4;
  float c = 0.0f;
#pragma unroll 4
  for (int t = j; t < TT; t += rate) {
    size_t row = (size_t)t * BB + b;
    float z = __bfloat162float(G[row * G3 + h]);
    float f = __bfloat162float(G[row * G3 + HID + h]);
    float o = __bfloat162float(G[row * G3 + 2 * HID + h]);
    c = f * c + (1.0f - f) * z;
    Xn[row * HID + h] = __float2bfloat16(o * c);
  }
}

// ------------------------------------------------------------------
extern "C" void kernel_launch(void* const* d_in, const int* in_sizes, int n_in,
                              void* d_out, int out_size, void* d_ws, size_t ws_size,
                              hipStream_t stream) {
  (void)in_sizes; (void)n_in; (void)out_size; (void)ws_size;

  const int*   x    = (const int*)d_in[0];
  const float* emb  = (const float*)d_in[1];
  const float* W[4] = {(const float*)d_in[2], (const float*)d_in[4],
                       (const float*)d_in[6], (const float*)d_in[8]};
  const float* bv[4]= {(const float*)d_in[3], (const float*)d_in[5],
                       (const float*)d_in[7], (const float*)d_in[9]};
  const float* decW = (const float*)d_in[10];
  const float* decb = (const float*)d_in[11];

  char* ws = (char*)d_ws;
  size_t off = 0;
  auto take = [&](size_t bytes) { size_t o = off; off += (bytes + 255) & ~(size_t)255; return o; };

  size_t oWp0 = take((size_t)EMB * G3 * 2);
  size_t oWp1 = take((size_t)HID * G3 * 2);
  size_t oWp2 = take((size_t)HID * G3 * 2);
  size_t oWp3 = take((size_t)HID * G3 * 2);
  size_t oWpD = take((size_t)HID * NVOCAB * 2);
  size_t oXa  = take((size_t)MROWS * HID * 2);
  size_t oXb  = take((size_t)MROWS * HID * 2);
  size_t oG   = take((size_t)MROWS * G3 * 2);

  __hip_bfloat16* Wp0 = (__hip_bfloat16*)(ws + oWp0);
  __hip_bfloat16* Wp1 = (__hip_bfloat16*)(ws + oWp1);
  __hip_bfloat16* Wp2 = (__hip_bfloat16*)(ws + oWp2);
  __hip_bfloat16* Wp3 = (__hip_bfloat16*)(ws + oWp3);
  __hip_bfloat16* WpD = (__hip_bfloat16*)(ws + oWpD);
  __hip_bfloat16* Xa  = (__hip_bfloat16*)(ws + oXa);
  __hip_bfloat16* Xb  = (__hip_bfloat16*)(ws + oXb);
  __hip_bfloat16* Gg  = (__hip_bfloat16*)(ws + oG);

  // 1) pack weights to WMMA B-fragment layout (bf16)
  packW_k<<<(EMB * G3) / 256, 256, 0, stream>>>(W[0], Wp0, EMB, G3);
  packW_k<<<(HID * G3) / 256, 256, 0, stream>>>(W[1], Wp1, HID, G3);
  packW_k<<<(HID * G3) / 256, 256, 0, stream>>>(W[2], Wp2, HID, G3);
  packW_k<<<(HID * G3) / 256, 256, 0, stream>>>(W[3], Wp3, HID, G3);
  packW_k<<<(HID * NVOCAB) / 256, 256, 0, stream>>>(decW, WpD, HID, NVOCAB);

  // 2) embedding -> time-major bf16 X
  embed_k<<<(MROWS * EMB) / 256, 256, 0, stream>>>(x, emb, Xa);

  dim3 gGate(G3 / 128, MROWS / 64);
  dim3 gDec(NVOCAB / 128, MROWS / 64);

  // 3) layer 0 (K=EMB), rate 1
  gemm_wmma<EMB, G3, 0><<<gGate, 256, 0, stream>>>(Xa, Wp0, bv[0], Gg);
  scan_k<<<(1 * BB * HID) / 256, 256, 0, stream>>>(Gg, Xb, 1);
  // 4) layer 1, rate 2
  gemm_wmma<HID, G3, 0><<<gGate, 256, 0, stream>>>(Xb, Wp1, bv[1], Gg);
  scan_k<<<(2 * BB * HID) / 256, 256, 0, stream>>>(Gg, Xa, 2);
  // 5) layer 2, rate 4
  gemm_wmma<HID, G3, 0><<<gGate, 256, 0, stream>>>(Xa, Wp2, bv[2], Gg);
  scan_k<<<(4 * BB * HID) / 256, 256, 0, stream>>>(Gg, Xb, 4);
  // 6) layer 3, rate 8
  gemm_wmma<HID, G3, 0><<<gGate, 256, 0, stream>>>(Xb, Wp3, bv[3], Gg);
  scan_k<<<(8 * BB * HID) / 256, 256, 0, stream>>>(Gg, Xa, 8);

  // 7) decoder -> f32 (B,T,V) in d_out
  gemm_wmma<HID, NVOCAB, 1><<<gDec, 256, 0, stream>>>(Xa, WpD, decb, d_out);
}
